// PathGraphFormer_11596411699779
// MI455X (gfx1250) — compile-verified, matrix operand order checked by reading, not run on hardware
//
#include <hip/hip_runtime.h>
#include <hip/hip_bf16.h>

// ---------------- model dims ----------------
#define NN_   3072
#define SS_   256
#define DIN_  64
#define DD_   256
#define HH_   8
#define LL_   3
#define DFF_  1024
#define FEAT_ 20
#define LN_EPS_ 1e-5f

typedef __attribute__((ext_vector_type(16))) __bf16 v16bf;
typedef __attribute__((ext_vector_type(8)))  __bf16 v8bf;
typedef __attribute__((ext_vector_type(8)))  float  v8f;

__device__ __forceinline__ v8f wmma_bf16(v16bf a, v16bf b, v8f c) {
  return __builtin_amdgcn_wmma_f32_16x16x32_bf16(false, a, false, b, (short)0, c,
                                                 false, false);
}

// Fragment load: per ISA 7.12.2, lane-half hf needs K runs {hf*8..hf*8+7} and
// {16+hf*8..16+hf*8+7} -> two contiguous 16B bf16 chunks -> two global_load_b128.
__device__ __forceinline__ v16bf load_frag(const __bf16* p, int hf) {
  v8bf c0 = *reinterpret_cast<const v8bf*>(p + hf * 8);
  v8bf c1 = *reinterpret_cast<const v8bf*>(p + 16 + hf * 8);
  return __builtin_shufflevector(c0, c1, 0, 1, 2, 3, 4, 5, 6, 7,
                                 8, 9, 10, 11, 12, 13, 14, 15);
}

__device__ __forceinline__ int kpat(int e, int hf) {
  return ((e >> 3) << 4) + (hf << 3) + (e & 7);
}

// =====================================================================
// Register-blocked GEMM: each wave computes a 32x64 tile (2x4 WMMA tiles).
// C = act( scale * A[M,K](bf16) @ Bt[Nt,K](bf16)^T + bias ) + residual
// Outputs: optional fp32 (outF) and/or bf16 (outB, optionally transposed).
// =====================================================================
__global__ __launch_bounds__(256) void gemm_bf16_kernel(
    const __bf16* __restrict__ A, const int* __restrict__ Aidx, int lda,
    const __bf16* __restrict__ Bt, int ldb,
    const float* __restrict__ bias,
    const float* __restrict__ scale_ptr,
    const float* __restrict__ residual, int ldr,
    float* __restrict__ outF, int ldf,
    __bf16* __restrict__ outB, int ldbo, int transB_out,
    int M, int Nt, int K, int act)
{
  const int lane = threadIdx.x & 31;
  const int wave = threadIdx.x >> 5;
  const int hf   = lane >> 4;
  const int lan  = lane & 15;

  const long ntn   = Nt >> 6;                 // 64-wide column blocks
  const long tile  = (long)blockIdx.x * 8 + wave;
  const long total = (long)(M >> 5) * ntn;    // 32-high row blocks
  if (tile >= total) return;                  // whole-wave exit, EXEC all-1s

  const int mt = (int)(tile / ntn);
  const int nt = (int)(tile % ntn);

  const __bf16* Ar[2];
#pragma unroll
  for (int mi = 0; mi < 2; ++mi) {
    int r = mt * 32 + mi * 16 + lan;
    if (Aidx) r = Aidx[r];
    Ar[mi] = A + (long)r * lda;
  }
  const __bf16* Br[4];
#pragma unroll
  for (int ni = 0; ni < 4; ++ni)
    Br[ni] = Bt + (long)(nt * 64 + ni * 16 + lan) * ldb;

  v8f acc[2][4];
#pragma unroll
  for (int mi = 0; mi < 2; ++mi)
#pragma unroll
    for (int ni = 0; ni < 4; ++ni)
#pragma unroll
      for (int i = 0; i < 8; ++i) acc[mi][ni][i] = 0.0f;

  for (int k0 = 0; k0 < K; k0 += 32) {
    if (k0 + 32 < K) {                        // CDNA5 global_prefetch_b8
      __builtin_prefetch(Ar[0] + k0 + 32, 0, 1);
      __builtin_prefetch(Br[0] + k0 + 32, 0, 1);
    }
    v16bf af[2], bfr[4];
#pragma unroll
    for (int mi = 0; mi < 2; ++mi) af[mi] = load_frag(Ar[mi] + k0, hf);
#pragma unroll
    for (int ni = 0; ni < 4; ++ni) bfr[ni] = load_frag(Br[ni] + k0, hf);
#pragma unroll
    for (int mi = 0; mi < 2; ++mi)
#pragma unroll
      for (int ni = 0; ni < 4; ++ni)
        acc[mi][ni] = wmma_bf16(af[mi], bfr[ni], acc[mi][ni]);
  }

  const float scale = scale_ptr ? *scale_ptr : 1.0f;
#pragma unroll
  for (int mi = 0; mi < 2; ++mi)
#pragma unroll
    for (int ni = 0; ni < 4; ++ni) {
      const int  ncol = nt * 64 + ni * 16 + lan;
      const float bv  = bias ? bias[ncol] : 0.0f;
#pragma unroll
      for (int r = 0; r < 8; ++r) {
        const long mrow = mt * 32 + mi * 16 + hf * 8 + r;
        float v = acc[mi][ni][r] * scale + bv;
        if (act == 1) v = 0.5f * v * (1.0f + erff(v * 0.70710678118f));
        if (residual) v += residual[mrow * ldr + ncol];
        if (outF) outF[mrow * ldf + ncol] = v;
        if (outB) {
          if (transB_out) outB[(long)ncol * ldbo + mrow] = (__bf16)v;
          else            outB[mrow * (long)ldbo + ncol] = (__bf16)v;
        }
      }
    }
}

// =====================================================================
// Fused biased MHA, flash-style online softmax. bf16 Q/K (row-major) and
// V^T (dim-major) operands -> all fragment loads are contiguous b128.
// bias(q,k) = bscale * biasPtr[q*brs + k*bcs] (fp32, nullable)
// =====================================================================
__global__ __launch_bounds__(256) void attn_bf16_kernel(
    const __bf16* __restrict__ Qm, int ldq,
    const __bf16* __restrict__ Km, int ldk,
    const __bf16* __restrict__ Vt, int ldvt,   // V^T: [DD_, Lk]
    int Lq, int Lk,
    const float* __restrict__ biasPtr, long brs, long bcs,
    const float* __restrict__ bscale_ptr,
    const unsigned char* __restrict__ pad,
    __bf16* __restrict__ Out)                  // [Lq, DD_] bf16
{
  __shared__ float pstage[8][16][33];

  const int lane = threadIdx.x & 31;
  const int wave = threadIdx.x >> 5;
  const int hf   = lane >> 4;
  const int lan  = lane & 15;

  const long task  = (long)blockIdx.x * 8 + wave;
  const long total = (long)(Lq >> 4) * HH_;
  if (task >= total) return;

  const int qt = (int)(task >> 3);
  const int hh = (int)(task & 7);

  const v16bf qf = load_frag(Qm + (long)(qt * 16 + lan) * ldq + hh * 32, hf);

  const float bscale = bscale_ptr ? *bscale_ptr : 1.0f;
  const float invsq  = 0.17677669529663687f;   // 1/sqrt(32)

  float m_run[8], l_run[8];
  v8f acc0, acc1;
#pragma unroll
  for (int r = 0; r < 8; ++r) {
    m_run[r] = -1e30f; l_run[r] = 0.0f; acc0[r] = 0.0f; acc1[r] = 0.0f;
  }

  const int nchunks = Lk >> 5;
  for (int ck = 0; ck < nchunks; ++ck) {
    const int kbase = ck * 32;
    float s[2][8];
#pragma unroll
    for (int sub = 0; sub < 2; ++sub) {
      const int keyrow = kbase + sub * 16 + lan;
      const v16bf kf = load_frag(Km + (long)keyrow * ldk + hh * 32, hf);
      v8f sc;
#pragma unroll
      for (int i = 0; i < 8; ++i) sc[i] = 0.0f;
      sc = wmma_bf16(qf, kf, sc);

      const float pd = (pad && pad[keyrow]) ? 10000.0f : 0.0f;
#pragma unroll
      for (int r = 0; r < 8; ++r) {
        const long qrow = qt * 16 + hf * 8 + r;
        float bv = biasPtr ? bscale * biasPtr[qrow * brs + (long)keyrow * bcs] : 0.0f;
        s[sub][r] = sc[r] * invsq + bv - pd;
      }
    }

    float scf[8], p0[8], p1[8];
#pragma unroll
    for (int r = 0; r < 8; ++r) {
      float v = fmaxf(s[0][r], s[1][r]);
      for (int off = 8; off >= 1; off >>= 1) v = fmaxf(v, __shfl_xor(v, off, 32));
      const float nm = fmaxf(m_run[r], v);
      scf[r] = __expf(m_run[r] - nm);
      m_run[r] = nm;
      p0[r] = __expf(s[0][r] - nm);
      p1[r] = __expf(s[1][r] - nm);
      float rs = p0[r] + p1[r];
      for (int off = 8; off >= 1; off >>= 1) rs += __shfl_xor(rs, off, 32);
      l_run[r] = l_run[r] * scf[r] + rs;
      acc0[r] *= scf[r];
      acc1[r] *= scf[r];
    }

    // P (C-layout) -> LDS -> bf16 A fragment (same wave; compiler inserts waits)
#pragma unroll
    for (int r = 0; r < 8; ++r) {
      pstage[wave][hf * 8 + r][lan]      = p0[r];
      pstage[wave][hf * 8 + r][16 + lan] = p1[r];
    }
    v16bf pf;
#pragma unroll
    for (int e = 0; e < 16; ++e) pf[e] = (__bf16)pstage[wave][lan][kpat(e, hf)];

    // P @ V with V^T fragments as contiguous runs over keys
#pragma unroll
    for (int dsub = 0; dsub < 2; ++dsub) {
      const int dcol = hh * 32 + dsub * 16 + lan;
      const v16bf vf = load_frag(Vt + (long)dcol * ldvt + kbase, hf);
      if (dsub == 0) acc0 = wmma_bf16(pf, vf, acc0);
      else           acc1 = wmma_bf16(pf, vf, acc1);
    }
  }

#pragma unroll
  for (int r = 0; r < 8; ++r) {
    const long qrow = qt * 16 + hf * 8 + r;
    const float inv = 1.0f / l_run[r];
    Out[qrow * DD_ + hh * 32 + lan]      = (__bf16)(acc0[r] * inv);
    Out[qrow * DD_ + hh * 32 + 16 + lan] = (__bf16)(acc1[r] * inv);
  }
}

// =====================================================================
// Block reductions
// =====================================================================
__device__ __forceinline__ float blk_red_add(float v, float* red) {
  const int t = threadIdx.x;
  red[t] = v; __syncthreads();
  for (int s = 128; s > 0; s >>= 1) { if (t < s) red[t] += red[t + s]; __syncthreads(); }
  const float r = red[0]; __syncthreads();
  return r;
}
__device__ __forceinline__ float blk_red_max(float v, float* red) {
  const int t = threadIdx.x;
  red[t] = v; __syncthreads();
  for (int s = 128; s > 0; s >>= 1) { if (t < s) red[t] = fmaxf(red[t], red[t + s]); __syncthreads(); }
  const float r = red[0]; __syncthreads();
  return r;
}
__device__ __forceinline__ float blk_red_min(float v, float* red) {
  const int t = threadIdx.x;
  red[t] = v; __syncthreads();
  for (int s = 128; s > 0; s >>= 1) { if (t < s) red[t] = fminf(red[t], red[t + s]); __syncthreads(); }
  const float r = red[0]; __syncthreads();
  return r;
}

// =====================================================================
// fp32 -> bf16 elementwise (weights/x staging)
// =====================================================================
__global__ __launch_bounds__(256) void cvt_bf16_kernel(
    const float* __restrict__ in, __bf16* __restrict__ out, long n)
{
  for (long i = (long)blockIdx.x * 256 + threadIdx.x; i < n;
       i += (long)gridDim.x * 256)
    out[i] = (__bf16)in[i];
}

// =====================================================================
// Per-node stats + RBF features z[N,20]; Cn(bf16) = C/(||C||+1e-6)
// =====================================================================
__global__ __launch_bounds__(256) void node_stats_kernel(
    const float* __restrict__ C, __bf16* __restrict__ Cn, float* __restrict__ z)
{
  __shared__ float red[256];
  const int row = blockIdx.x, t = threadIdx.x;
  const float c  = C[(long)row * SS_ + t];
  const float cc = fminf(fmaxf(c, 0.0f), 1.0f);

  const float sum   = blk_red_add(cc, red);
  const float cmax  = blk_red_max(cc, red);
  const float cmin  = blk_red_min(cc, red);
  const float sumsq = blk_red_add(cc * cc, red);
  const float nrm2  = blk_red_add(c * c, red);

  const float cm   = sum / (float)SS_;
  const float cstd = sqrtf(fmaxf(sumsq / (float)SS_ - cm * cm, 0.0f));
  const float inrm = 1.0f / (sqrtf(nrm2) + 1e-6f);

  Cn[(long)row * SS_ + t] = (__bf16)(c * inrm);

  if (t < FEAT_) {
    float v;
    if (t == 0)      v = cm;
    else if (t == 1) v = cmax;
    else if (t == 2) v = cmin;
    else if (t == 3) v = cstd;
    else {
      const int   j   = (t - 4) & 7;
      const float val = (t < 12) ? cm : cmax;
      const float cj  = (float)j * (1.0f / 7.0f);
      const float u   = (val - cj) / 0.150001f;
      v = __expf(-0.5f * u * u);
    }
    z[(long)row * FEAT_ + t] = v;
  }
}

// =====================================================================
// h += gate * LN(z @ pe_W^T + pe_b)  (masked); also refresh h_bf
// =====================================================================
__global__ __launch_bounds__(256) void pe_add_kernel(
    const float* __restrict__ z, const float* __restrict__ pe_W,
    const float* __restrict__ pe_b, const float* __restrict__ pe_g,
    const float* __restrict__ pe_beta, const float* __restrict__ gate_p,
    const unsigned char* __restrict__ mask,
    float* __restrict__ h, __bf16* __restrict__ h_bf)
{
  __shared__ float zl[FEAT_];
  __shared__ float red[256];
  const int row = blockIdx.x, t = threadIdx.x;
  if (t < FEAT_) zl[t] = z[(long)row * FEAT_ + t];
  __syncthreads();

  float v = pe_b[t];
#pragma unroll
  for (int f = 0; f < FEAT_; ++f) v += zl[f] * pe_W[t * FEAT_ + f];

  const float mean = blk_red_add(v, red) / (float)DD_;
  const float d    = v - mean;
  const float var  = blk_red_add(d * d, red) / (float)DD_;

  float pe = (d * rsqrtf(var + LN_EPS_) * pe_g[t] + pe_beta[t]) * (*gate_p);
  if (mask[row]) pe = 0.0f;
  const float hv = h[(long)row * DD_ + t] + pe;
  h[(long)row * DD_ + t]    = hv;
  h_bf[(long)row * DD_ + t] = (__bf16)hv;
}

// =====================================================================
// LayerNorm over D=256; fp32 and/or bf16 outputs.
// =====================================================================
__global__ __launch_bounds__(256) void layernorm_kernel(
    const float* __restrict__ x, const float* __restrict__ g,
    const float* __restrict__ b, float* __restrict__ outF,
    __bf16* __restrict__ outB)
{
  __shared__ float red[256];
  const int row = blockIdx.x, t = threadIdx.x;
  const float v    = x[(long)row * DD_ + t];
  const float mean = blk_red_add(v, red) / (float)DD_;
  const float d    = v - mean;
  const float var  = blk_red_add(d * d, red) / (float)DD_;
  const float o    = d * rsqrtf(var + LN_EPS_) * g[t] + b[t];
  if (outF) outF[(long)row * DD_ + t] = o;
  if (outB) outB[(long)row * DD_ + t] = (__bf16)o;
}

// =====================================================================
// Host orchestration
// =====================================================================
static inline void launch_gemm(hipStream_t s,
    const __bf16* A, const int* Aidx, int lda,
    const __bf16* Bt, int ldb, const float* bias, const float* scale,
    const float* res, int ldr,
    float* outF, int ldf, __bf16* outB, int ldbo, int transB,
    int M, int Nt, int K, int act)
{
  long tiles = (long)(M / 32) * (Nt / 64);
  int blocks = (int)((tiles + 7) / 8);
  gemm_bf16_kernel<<<blocks, 256, 0, s>>>(A, Aidx, lda, Bt, ldb, bias, scale,
                                          res, ldr, outF, ldf, outB, ldbo,
                                          transB, M, Nt, K, act);
}

static inline void launch_attn(hipStream_t s,
    const __bf16* Q, int ldq, const __bf16* K, int ldk,
    const __bf16* Vt, int ldvt,
    int Lq, int Lk, const float* biasPtr, long brs, long bcs,
    const float* bscale, const unsigned char* pad, __bf16* Out)
{
  long tasks = (long)(Lq / 16) * HH_;
  int blocks = (int)((tasks + 7) / 8);
  attn_bf16_kernel<<<blocks, 256, 0, s>>>(Q, ldq, K, ldk, Vt, ldvt, Lq, Lk,
                                          biasPtr, brs, bcs, bscale, pad, Out);
}

static inline void launch_cvt(hipStream_t s, const float* in, __bf16* out, long n)
{
  int blocks = (int)((n + 255) / 256);
  if (blocks > 2048) blocks = 2048;
  cvt_bf16_kernel<<<blocks, 256, 0, s>>>(in, out, n);
}

extern "C" void kernel_launch(void* const* d_in, const int* in_sizes, int n_in,
                              void* d_out, int out_size, void* d_ws, size_t ws_size,
                              hipStream_t stream)
{
  (void)in_sizes; (void)n_in; (void)out_size; (void)ws_size;

  const float* x        = (const float*)d_in[0];
  const float* C        = (const float*)d_in[1];
  const int*   sink_idx = (const int*)d_in[2];
  const unsigned char* mask = (const unsigned char*)d_in[3];
  const float* proj_in_W = (const float*)d_in[4];
  const float* proj_in_b = (const float*)d_in[5];
  const float* pe_W    = (const float*)d_in[6];
  const float* pe_b    = (const float*)d_in[7];
  const float* pe_g    = (const float*)d_in[8];
  const float* pe_beta = (const float*)d_in[9];
  const float* pe_gate = (const float*)d_in[10];
  const float* alpha   = (const float*)d_in[11];
  const float* beta    = (const float*)d_in[12];
  const float* sq_W    = (const float*)d_in[13];
  const float* sq_b    = (const float*)d_in[14];
  const float* sn_Win  = (const float*)d_in[15];
  const float* sn_bin  = (const float*)d_in[16];
  const float* sn_Wout = (const float*)d_in[17];
  const float* sn_bout = (const float*)d_in[18];
  const float* nn_Win  = (const float*)d_in[19];
  const float* nn_bin  = (const float*)d_in[20];
  const float* nn_Wout = (const float*)d_in[21];
  const float* nn_bout = (const float*)d_in[22];
  const float* ln_s_g  = (const float*)d_in[23];
  const float* ln_s_b  = (const float*)d_in[24];
  const float* ln_n1_g = (const float*)d_in[25];
  const float* ln_n1_b = (const float*)d_in[26];
  const float* ln_n2_g = (const float*)d_in[27];
  const float* ln_n2_b = (const float*)d_in[28];
  const float* ffn_W1  = (const float*)d_in[29];
  const float* ffn_b1  = (const float*)d_in[30];
  const float* ffn_W2  = (const float*)d_in[31];
  const float* ffn_b2  = (const float*)d_in[32];
  const float* ffs_W1  = (const float*)d_in[33];
  const float* ffs_b1  = (const float*)d_in[34];
  const float* ffs_W2  = (const float*)d_in[35];
  const float* ffs_b2  = (const float*)d_in[36];
  const float* ln_out_g = (const float*)d_in[37];
  const float* ln_out_b = (const float*)d_in[38];

  // ---------------- workspace layout (128B aligned slabs) ----------------
  char* base = (char*)d_ws;
  auto alloc = [&](size_t bytes) -> void* {
    void* p = (void*)base;
    base += (bytes + 127) & ~(size_t)127;
    return p;
  };
  // fp32 masters
  float* h       = (float*)alloc(sizeof(float) * NN_ * DD_);
  float* sink_q  = (float*)alloc(sizeof(float) * SS_ * DD_);
  float* bias_nn = (float*)alloc(sizeof(float) * (size_t)NN_ * NN_);
  float* z       = (float*)alloc(sizeof(float) * NN_ * FEAT_);
  // bf16 operands
  __bf16* h_bf      = (__bf16*)alloc(2ul * NN_ * DD_);
  __bf16* hn_bf     = (__bf16*)alloc(2ul * NN_ * DD_);
  __bf16* sinkq_bf  = (__bf16*)alloc(2ul * SS_ * DD_);
  __bf16* sqn_bf    = (__bf16*)alloc(2ul * SS_ * DD_);
  __bf16* qs_bf     = (__bf16*)alloc(2ul * SS_ * DD_);
  __bf16* sattn_bf  = (__bf16*)alloc(2ul * SS_ * DD_);
  __bf16* qk_bf     = (__bf16*)alloc(2ul * NN_ * 2 * DD_);   // node q|k / sink k
  __bf16* vT_bf     = (__bf16*)alloc(2ul * DD_ * NN_);        // V^T
  __bf16* attnout_bf= (__bf16*)alloc(2ul * NN_ * DD_);
  __bf16* ffmid_bf  = (__bf16*)alloc(2ul * NN_ * DFF_);
  __bf16* sffmid_bf = (__bf16*)alloc(2ul * SS_ * DFF_);
  __bf16* Cn_bf     = (__bf16*)alloc(2ul * NN_ * SS_);
  __bf16* x_bf      = (__bf16*)alloc(2ul * NN_ * DIN_);
  // bf16 weights
  __bf16* pw_bf     = (__bf16*)alloc(2ul * DD_ * DIN_);
  __bf16* sqw_bf    = (__bf16*)alloc(2ul * DD_ * DD_);
  __bf16* snWin_bf  = (__bf16*)alloc(2ul * LL_ * 3 * DD_ * DD_);
  __bf16* snWout_bf = (__bf16*)alloc(2ul * LL_ * DD_ * DD_);
  __bf16* nnWin_bf  = (__bf16*)alloc(2ul * LL_ * 3 * DD_ * DD_);
  __bf16* nnWout_bf = (__bf16*)alloc(2ul * LL_ * DD_ * DD_);
  __bf16* ffnW1_bf  = (__bf16*)alloc(2ul * LL_ * DFF_ * DD_);
  __bf16* ffnW2_bf  = (__bf16*)alloc(2ul * LL_ * DD_ * DFF_);
  __bf16* ffsW1_bf  = (__bf16*)alloc(2ul * LL_ * DFF_ * DD_);
  __bf16* ffsW2_bf  = (__bf16*)alloc(2ul * LL_ * DD_ * DFF_);

  // ---- stage weights & x to bf16
  launch_cvt(stream, x, x_bf, (long)NN_ * DIN_);
  launch_cvt(stream, proj_in_W, pw_bf, (long)DD_ * DIN_);
  launch_cvt(stream, sq_W, sqw_bf, (long)DD_ * DD_);
  launch_cvt(stream, sn_Win, snWin_bf, (long)LL_ * 3 * DD_ * DD_);
  launch_cvt(stream, sn_Wout, snWout_bf, (long)LL_ * DD_ * DD_);
  launch_cvt(stream, nn_Win, nnWin_bf, (long)LL_ * 3 * DD_ * DD_);
  launch_cvt(stream, nn_Wout, nnWout_bf, (long)LL_ * DD_ * DD_);
  launch_cvt(stream, ffn_W1, ffnW1_bf, (long)LL_ * DFF_ * DD_);
  launch_cvt(stream, ffn_W2, ffnW2_bf, (long)LL_ * DD_ * DFF_);
  launch_cvt(stream, ffs_W1, ffsW1_bf, (long)LL_ * DFF_ * DD_);
  launch_cvt(stream, ffs_W2, ffsW2_bf, (long)LL_ * DD_ * DFF_);

  // ---- h = x @ proj_in_W^T + b (fp32 master; h_bf refreshed by pe_add)
  launch_gemm(stream, x_bf, nullptr, DIN_, pw_bf, DIN_, proj_in_b, nullptr,
              nullptr, 0, h, DD_, nullptr, 0, 0, NN_, DD_, DIN_, 0);
  // ---- stats: Cn(bf16), z
  node_stats_kernel<<<NN_, 256, 0, stream>>>(C, Cn_bf, z);
  // ---- h += pe ; h_bf = bf16(h)
  pe_add_kernel<<<NN_, 256, 0, stream>>>(z, pe_W, pe_b, pe_g, pe_beta, pe_gate,
                                         mask, h, h_bf);
  // ---- bias_nn = alpha * Cn @ Cn^T (fp32)
  launch_gemm(stream, Cn_bf, nullptr, SS_, Cn_bf, SS_, nullptr, alpha,
              nullptr, 0, bias_nn, NN_, nullptr, 0, 0, NN_, NN_, SS_, 0);
  // ---- sink_q = h[sink_idx] @ sq_W^T + sq_b (fp32 master)
  launch_gemm(stream, h_bf, sink_idx, DD_, sqw_bf, DD_, sq_b, nullptr,
              nullptr, 0, sink_q, DD_, nullptr, 0, 0, SS_, DD_, DD_, 0);

  for (int l = 0; l < LL_; ++l) {
    const __bf16* Win_s  = snWin_bf  + (size_t)l * 3 * DD_ * DD_;
    const float*  bin_s  = sn_bin    + (size_t)l * 3 * DD_;
    const __bf16* Wout_s = snWout_bf + (size_t)l * DD_ * DD_;
    const float*  bout_s = sn_bout   + (size_t)l * DD_;
    const __bf16* Win_n  = nnWin_bf  + (size_t)l * 3 * DD_ * DD_;
    const float*  bin_n  = nn_bin    + (size_t)l * 3 * DD_;
    const __bf16* Wout_n = nnWout_bf + (size_t)l * DD_ * DD_;
    const float*  bout_n = nn_bout   + (size_t)l * DD_;

    // ---- sink branch ----
    layernorm_kernel<<<SS_, 256, 0, stream>>>(sink_q, ln_s_g + l * DD_,
                                              ln_s_b + l * DD_, nullptr, sqn_bf);
    // q_s = sqn @ Win[0:D]^T + bin[0:D]
    launch_gemm(stream, sqn_bf, nullptr, DD_, Win_s, DD_, bin_s, nullptr,
                nullptr, 0, nullptr, 0, qs_bf, DD_, 0, SS_, DD_, DD_, 0);
    // k_s = h @ Win[D:2D]^T + bin[D:2D]
    launch_gemm(stream, h_bf, nullptr, DD_, Win_s + DD_ * DD_, DD_, bin_s + DD_,
                nullptr, nullptr, 0, nullptr, 0, qk_bf, DD_, 0, NN_, DD_, DD_, 0);
    // v_s^T = (h @ Win[2D:3D]^T + bin[2D:3D])^T  -> vT_bf [DD_, NN_]
    launch_gemm(stream, h_bf, nullptr, DD_, Win_s + 2 * DD_ * DD_, DD_,
                bin_s + 2 * DD_, nullptr, nullptr, 0, nullptr, 0,
                vT_bf, NN_, 1, NN_, DD_, DD_, 0);
    // attention: bias(q,k) = beta * C[k, q]
    launch_attn(stream, qs_bf, DD_, qk_bf, DD_, vT_bf, NN_,
                SS_, NN_, C, 1, SS_, beta, mask, sattn_bf);
    // sink_q += sattn @ Wout^T + bout  (fp32 master + bf16 for FF1)
    launch_gemm(stream, sattn_bf, nullptr, DD_, Wout_s, DD_, bout_s, nullptr,
                sink_q, DD_, sink_q, DD_, sinkq_bf, DD_, 0, SS_, DD_, DD_, 0);
    // sink FFN
    launch_gemm(stream, sinkq_bf, nullptr, DD_, ffsW1_bf + (size_t)l * DFF_ * DD_,
                DD_, ffs_b1 + (size_t)l * DFF_, nullptr, nullptr, 0,
                nullptr, 0, sffmid_bf, DFF_, 0, SS_, DFF_, DD_, 1);
    launch_gemm(stream, sffmid_bf, nullptr, DFF_, ffsW2_bf + (size_t)l * DD_ * DFF_,
                DFF_, ffs_b2 + (size_t)l * DD_, nullptr, sink_q, DD_,
                sink_q, DD_, nullptr, 0, 0, SS_, DD_, DFF_, 0);

    // ---- node branch ----
    layernorm_kernel<<<NN_, 256, 0, stream>>>(h, ln_n1_g + l * DD_,
                                              ln_n1_b + l * DD_, nullptr, hn_bf);
    // q|k = hn @ Win[0:2D]^T + bin[0:2D] -> qk_bf [N, 512]
    launch_gemm(stream, hn_bf, nullptr, DD_, Win_n, DD_, bin_n, nullptr,
                nullptr, 0, nullptr, 0, qk_bf, 2 * DD_, 0, NN_, 2 * DD_, DD_, 0);
    // v^T = (hn @ Win[2D:3D]^T + bin[2D:3D])^T
    launch_gemm(stream, hn_bf, nullptr, DD_, Win_n + 2 * DD_ * DD_, DD_,
                bin_n + 2 * DD_, nullptr, nullptr, 0, nullptr, 0,
                vT_bf, NN_, 1, NN_, DD_, DD_, 0);
    // attention with materialized bias_nn
    launch_attn(stream, qk_bf, 2 * DD_, qk_bf + DD_, 2 * DD_, vT_bf, NN_,
                NN_, NN_, bias_nn, NN_, 1, nullptr, mask, attnout_bf);
    // h += attnout @ Wout^T + bout (fp32 master only; h_bf refreshed by FF2)
    launch_gemm(stream, attnout_bf, nullptr, DD_, Wout_n, DD_, bout_n, nullptr,
                h, DD_, h, DD_, nullptr, 0, 0, NN_, DD_, DD_, 0);
    // node FFN on LN(h)
    layernorm_kernel<<<NN_, 256, 0, stream>>>(h, ln_n2_g + l * DD_,
                                              ln_n2_b + l * DD_, nullptr, hn_bf);
    launch_gemm(stream, hn_bf, nullptr, DD_, ffnW1_bf + (size_t)l * DFF_ * DD_,
                DD_, ffn_b1 + (size_t)l * DFF_, nullptr, nullptr, 0,
                nullptr, 0, ffmid_bf, DFF_, 0, NN_, DFF_, DD_, 1);
    // h += ffmid @ W2^T + b2 ; refresh h_bf for next layer's sink K/V
    launch_gemm(stream, ffmid_bf, nullptr, DFF_, ffnW2_bf + (size_t)l * DD_ * DFF_,
                DFF_, ffn_b2 + (size_t)l * DD_, nullptr, h, DD_,
                h, DD_, h_bf, DD_, 0, NN_, DD_, DFF_, 0);
  }

  // ---- out = LN(sink_q) (fp32)
  layernorm_kernel<<<SS_, 256, 0, stream>>>(sink_q, ln_out_g, ln_out_b,
                                            (float*)d_out, nullptr);
}
